// monkey_switch_group_newQKV_status5_62929860821002
// MI455X (gfx1250) — matrix operand order repaired and checked by reading, not compiled, not run on hardware
//
#include <hip/hip_runtime.h>
#include <hip/hip_bf16.h>

// ---------------------------------------------------------------------------
// Types for CDNA5 WMMA (wave32, v_wmma_f32_16x16x32_bf16)
// ---------------------------------------------------------------------------
typedef __attribute__((ext_vector_type(16))) __bf16 bf16x16;
typedef __attribute__((ext_vector_type(8)))  float  floatx8;

union Frag {
    bf16x16        v;
    uint4          q[2];
    unsigned short u[16];
};

__device__ __forceinline__ unsigned short f2bf(float f) {
    union { float f; unsigned int u; } c; c.f = f;
    unsigned int u = c.u;
    u += 0x7FFFu + ((u >> 16) & 1u);   // round-to-nearest-even
    return (unsigned short)(u >> 16);
}

__device__ __forceinline__ floatx8 wmma_bf16(const Frag& a, const Frag& b, floatx8 c) {
    return __builtin_amdgcn_wmma_f32_16x16x32_bf16(false, a.v, false, b.v,
                                                   (short)0, c, false, false);
}

// ---------------------------------------------------------------------------
// Problem constants
// ---------------------------------------------------------------------------
#define NB      256          // batch
#define HIDC    512          // hidden channels
#define HW      49           // 7x7
#define NGRP    16
#define GRPN    16           // 256/16
#define M_TOT   (NB * HW)    // 12544 GEMM rows
#define FEAT_ELEMS ((size_t)NB * HIDC * HW)   // 6,422,528

// ---------------------------------------------------------------------------
// Layout-transform kernels (one-shot, bandwidth-trivial)
// ---------------------------------------------------------------------------
// (NB, C, 49) f32  ->  (NB, 49, C) bf16   (channel-last activations)
__global__ void permute_cl_bf16(const float* __restrict__ src,
                                unsigned short* __restrict__ dst,
                                int C, int total) {
    int i = blockIdx.x * 256 + threadIdx.x;
    if (i >= total) return;
    int ci = i % C;
    int r  = i / C;
    int p  = r % HW;
    int n  = r / HW;
    dst[i] = f2bf(src[((size_t)n * C + ci) * HW + p]);
}

// (512, Cin, 9) f32  ->  dst[co][t*Cin + ci] bf16   (tap-major weights)
__global__ void permute_w_bf16(const float* __restrict__ src,
                               unsigned short* __restrict__ dst,
                               int Cin, int total) {
    int i = blockIdx.x * 256 + threadIdx.x;
    if (i >= total) return;
    int t  = i % 9;
    int ci = (i / 9) % Cin;
    int co = i / (9 * Cin);
    dst[(size_t)co * Cin * 9 + t * Cin + ci] = f2bf(src[i]);
}

// ---------------------------------------------------------------------------
// Implicit-GEMM 3x3 pad-1 conv, bf16 WMMA, f32 accumulate.
//   input : (NB, 49, CIN)    bf16 channel-last
//   weight: (512, 9*CIN)     bf16 tap-major (k = t*CIN + ci)
//   M = NB*49, N = 512, K = CIN*9
// Tile 128(M) x 64(N), 256 threads = 8 waves (4 M x 2 N).
// Each wave: 32x32 patch = 2 A-frags x 2 B-frags -> 4 WMMAs (2 ds_b128/WMMA).
// Software pipelined: next global tiles fetched into regs during WMMA.
// ---------------------------------------------------------------------------
template <int CIN>
__global__ void __launch_bounds__(256)
conv3x3_wmma(const unsigned short* __restrict__ inBf,
             const unsigned short* __restrict__ wBf,
             const float* __restrict__ residual,    // optional
             float* __restrict__ outF,              // optional
             unsigned short* __restrict__ outBfCL)  // optional, channel-last
{
    constexpr int K = CIN * 9;
    __shared__ unsigned short lA[128][40];  // 80B stride: 16B aligned, conflict-free
    __shared__ unsigned short lB[64][40];

    const int tid  = threadIdx.x;
    const int lane = tid & 31;
    const int wave = tid >> 5;
    const int l    = lane & 15;
    const int hi   = lane >> 4;
    const int wm   = (wave & 3) * 32;   // wave row offset in 128-tall tile
    const int wn   = (wave >> 2) * 32;  // wave col offset in 64-wide tile
    const int mT   = blockIdx.x * 128;
    const int nT   = blockIdx.y * 64;

    // cooperative-load coordinates: 16B chunks; A: rows ldRow & ldRow+64, B: ldRow
    const int ldRow = tid >> 2;         // 0..63
    const int ldCh  = (tid & 3) * 8;

    // im2col row decode for the two A rows this thread stages
    const int mIdx0 = mT + ldRow;
    const int nImg0 = mIdx0 / HW;
    const int p0    = mIdx0 - nImg0 * HW;
    const int y0    = p0 / 7, x0 = p0 - (p0 / 7) * 7;
    const int mIdx1 = mT + ldRow + 64;
    const int nImg1 = mIdx1 / HW;
    const int p1    = mIdx1 - nImg1 * HW;
    const int y1    = p1 / 7, x1 = p1 - (p1 / 7) * 7;
    const unsigned short* inBase0 = inBf + (size_t)nImg0 * HW * CIN;
    const unsigned short* inBase1 = inBf + (size_t)nImg1 * HW * CIN;
    const unsigned short* wRow    = wBf + (size_t)(nT + ldRow) * K;

    floatx8 acc00 = {}, acc01 = {}, acc10 = {}, acc11 = {};

    // fetch two A chunks + one B chunk (16B each) for k-tile at kT
    auto fetch = [&](int kT, uint4& ra0, uint4& ra1, uint4& rb) {
        rb = *(const uint4*)&wRow[kT + ldCh];
        int kk  = kT + ldCh;
        int t   = kk / CIN;              // constant division (folds to mul-shift)
        int ci0 = kk - t * CIN;          // chunk shares one tap (CIN % 8 == 0)
        int dy  = t / 3 - 1;
        int dx  = t - (t / 3) * 3 - 1;
        uint4 z; z.x = z.y = z.z = z.w = 0u;
        ra0 = z; ra1 = z;
        int yy0 = y0 + dy, xx0 = x0 + dx;
        int yy1 = y1 + dy, xx1 = x1 + dx;
        if ((unsigned)yy0 < 7u && (unsigned)xx0 < 7u)
            ra0 = *(const uint4*)&inBase0[(size_t)(yy0 * 7 + xx0) * CIN + ci0];
        if ((unsigned)yy1 < 7u && (unsigned)xx1 < 7u)
            ra1 = *(const uint4*)&inBase1[(size_t)(yy1 * 7 + xx1) * CIN + ci0];
    };

    uint4 ra0, ra1, rb;
    fetch(0, ra0, ra1, rb);

    for (int kT = 0; kT < K; kT += 32) {
        *(uint4*)&lB[ldRow][ldCh]      = rb;
        *(uint4*)&lA[ldRow][ldCh]      = ra0;
        *(uint4*)&lA[ldRow + 64][ldCh] = ra1;
        __syncthreads();

        if (kT + 32 < K) {
            fetch(kT + 32, ra0, ra1, rb);   // overlaps with WMMA below
            __builtin_prefetch(&wRow[kT + 64 + ldCh], 0, 3);
        }

        Frag a0, a1, b0, b1;
        a0.q[0] = *(const uint4*)&lA[wm + l][hi * 8];
        a0.q[1] = *(const uint4*)&lA[wm + l][hi * 8 + 16];
        a1.q[0] = *(const uint4*)&lA[wm + 16 + l][hi * 8];
        a1.q[1] = *(const uint4*)&lA[wm + 16 + l][hi * 8 + 16];
        b0.q[0] = *(const uint4*)&lB[wn + l][hi * 8];
        b0.q[1] = *(const uint4*)&lB[wn + l][hi * 8 + 16];
        b1.q[0] = *(const uint4*)&lB[wn + 16 + l][hi * 8];
        b1.q[1] = *(const uint4*)&lB[wn + 16 + l][hi * 8 + 16];

        acc00 = wmma_bf16(a0, b0, acc00);
        acc01 = wmma_bf16(a0, b1, acc01);
        acc10 = wmma_bf16(a1, b0, acc10);
        acc11 = wmma_bf16(a1, b1, acc11);
        __syncthreads();
    }

    // epilogue: D layout: VGPR r -> M=r (lanes 0-15) / M=r+8 (lanes 16-31), N=l
    #pragma unroll
    for (int im = 0; im < 2; ++im) {
        floatx8 accL = im ? acc10 : acc00;
        floatx8 accR = im ? acc11 : acc01;
        const int c0 = nT + wn + l;
        const int c1 = c0 + 16;
        #pragma unroll
        for (int r = 0; r < 8; ++r) {
            int m   = mT + wm + im * 16 + r + hi * 8;
            int ni  = m / HW;
            int pp  = m - ni * HW;
            size_t o0 = ((size_t)ni * HIDC + c0) * HW + pp;     // NCHW f32 out
            size_t o1 = ((size_t)ni * HIDC + c1) * HW + pp;
            float v0 = accL[r], v1 = accR[r];
            if (residual) { v0 += residual[o0]; v1 += residual[o1]; }
            if (outF)  { outF[o0] = v0; outF[o1] = v1; }
            if (outBfCL) {                                      // channel-last bf16
                size_t b0i = ((size_t)ni * HW + pp) * HIDC + c0;
                size_t b1i = ((size_t)ni * HW + pp) * HIDC + c1;
                outBfCL[b0i] = f2bf(v0);
                outBfCL[b1i] = f2bf(v1);
            }
        }
    }
}

// ---------------------------------------------------------------------------
// q-conv: s[n][c] (c<5) spatially constant -> direct 45-MAC kernel.
// ---------------------------------------------------------------------------
__global__ void status_conv(const float* __restrict__ status, int statusCol,
                            const float* __restrict__ rois,
                            const float* __restrict__ wq,   // (512,5,3,3) f32
                            float* __restrict__ out) {
    int idx = blockIdx.x * 256 + threadIdx.x;
    if (idx >= NB * HIDC * HW) return;
    int pp = idx % HW;
    int co = (idx / HW) % HIDC;
    int n  = idx / (HIDC * HW);
    int y = pp / 7, x = pp - y * 7;

    float s[5];
    s[0] = status[n * 2 + statusCol];
    #pragma unroll
    for (int j = 0; j < 4; ++j) s[j + 1] = rois[n * 5 + 1 + j];

    float acc = 0.f;
    #pragma unroll
    for (int c = 0; c < 5; ++c) {
        #pragma unroll
        for (int ky = 0; ky < 3; ++ky) {
            #pragma unroll
            for (int kx = 0; kx < 3; ++kx) {
                int yy = y + ky - 1, xx = x + kx - 1;
                if ((unsigned)yy < 7u && (unsigned)xx < 7u)
                    acc += s[c] * wq[((co * 5 + c) * 3 + ky) * 3 + kx];
            }
        }
    }
    out[idx] = acc;
}

// ---------------------------------------------------------------------------
// Attention per (group g, pixel hw):
//   att = softmax( scale * q k^T )  via WMMA (K=512)
//   virt = att v                    via WMMA (K=16 padded to 32)
// ---------------------------------------------------------------------------
__global__ void __launch_bounds__(256)
attention_kernel(const float* __restrict__ q,
                 const float* __restrict__ k,
                 const float* __restrict__ v,
                 float* __restrict__ virt) {
    __shared__ unsigned short qs[16][520];
    __shared__ unsigned short ks[16][520];
    __shared__ unsigned short vs[16][520];
    __shared__ float          attS[16][17];
    __shared__ unsigned short attB[16][40];

    const int hw = blockIdx.x;
    const int g  = blockIdx.y;
    const int tid  = threadIdx.x;
    const int lane = tid & 31;
    const int wave = tid >> 5;
    const int l    = lane & 15;
    const int hi   = lane >> 4;
    const float scale = 0.044194173824159216f;  // 1/sqrt(512)

    for (int i = tid; i < 16 * HIDC; i += 256) {
        int row = i >> 9;
        int c   = i & 511;
        size_t off = ((size_t)(g * GRPN + row) * HIDC + c) * HW + hw;
        qs[row][c] = f2bf(q[off]);
        ks[row][c] = f2bf(k[off]);
        vs[row][c] = f2bf(v[off]);
    }
    __syncthreads();

    if (wave == 0) {   // EXEC all-1 within the wave
        floatx8 accA = {};
        #pragma unroll
        for (int kt = 0; kt < HIDC; kt += 32) {
            Frag a, b;
            a.q[0] = *(const uint4*)&qs[l][kt + hi * 8];
            a.q[1] = *(const uint4*)&qs[l][kt + hi * 8 + 16];
            b.q[0] = *(const uint4*)&ks[l][kt + hi * 8];
            b.q[1] = *(const uint4*)&ks[l][kt + hi * 8 + 16];
            accA = wmma_bf16(a, b, accA);
        }
        #pragma unroll
        for (int r = 0; r < 8; ++r)
            attS[r + hi * 8][l] = accA[r];
    }
    __syncthreads();

    if (tid < 16) {
        int i = tid;
        float mx = -3.4e38f;
        float e[16];
        #pragma unroll
        for (int j = 0; j < 16; ++j) {
            float a = attS[i][j] * scale;
            e[j] = a;
            mx = fmaxf(mx, a);
        }
        float s = 0.f;
        #pragma unroll
        for (int j = 0; j < 16; ++j) { e[j] = __expf(e[j] - mx); s += e[j]; }
        float inv = 1.f / s;
        #pragma unroll
        for (int j = 0; j < 16; ++j) attB[i][j] = f2bf(e[j] * inv);
        #pragma unroll
        for (int j = 16; j < 32; ++j) attB[i][j] = 0;
    }
    __syncthreads();

    Frag a2;
    a2.q[0] = *(const uint4*)&attB[l][hi * 16];
    a2.q[1] = *(const uint4*)&attB[l][hi * 16 + 8];

    #pragma unroll
    for (int t = 0; t < 4; ++t) {
        int c0 = (wave + t * 8) * 16;
        Frag b2;
        #pragma unroll
        for (int e = 0; e < 16; ++e)   // lanes hi=1 cover padded K=16..31 -> zeros
            b2.u[e] = hi ? (unsigned short)0 : vs[e][c0 + l];
        floatx8 accV = {};
        accV = wmma_bf16(a2, b2, accV);
        #pragma unroll
        for (int r = 0; r < 8; ++r) {
            int i = r + hi * 8;
            int c = c0 + l;
            virt[((size_t)(g * GRPN + i) * HIDC + c) * HW + hw] = accV[r];
        }
    }
}

// ---------------------------------------------------------------------------
// Per-sample norm (mean/var over C*H*W) + affine + ReLU -> bf16 channel-last
// ---------------------------------------------------------------------------
__global__ void __launch_bounds__(256)
groupnorm_relu(const float* __restrict__ virt,
               const float* __restrict__ gamma,
               const float* __restrict__ beta,
               unsigned short* __restrict__ hbCL) {
    __shared__ float r1[256], r2[256];
    const int n   = blockIdx.x;
    const int tid = threadIdx.x;
    const size_t base = (size_t)n * HIDC * HW;
    const int cnt = HIDC * HW;  // 25088

    float s = 0.f, s2 = 0.f;
    for (int i = tid; i < cnt; i += 256) {
        float xv = virt[base + i];
        s += xv; s2 += xv * xv;
    }
    r1[tid] = s; r2[tid] = s2;
    __syncthreads();
    for (int st = 128; st > 0; st >>= 1) {
        if (tid < st) { r1[tid] += r1[tid + st]; r2[tid] += r2[tid + st]; }
        __syncthreads();
    }
    const float inv = 1.f / (float)cnt;
    const float mu  = r1[0] * inv;
    const float var = r2[0] * inv - mu * mu;
    const float rs  = rsqrtf(var + 1e-5f);

    for (int i = tid; i < cnt; i += 256) {
        int c = i / HW;
        int pp = i - c * HW;
        float xv = (virt[base + i] - mu) * rs * gamma[c] + beta[c];
        hbCL[((size_t)n * HW + pp) * HIDC + c] = f2bf(fmaxf(xv, 0.f));
    }
}

// ---------------------------------------------------------------------------
// Launch
// ---------------------------------------------------------------------------
extern "C" void kernel_launch(void* const* d_in, const int* in_sizes, int n_in,
                              void* d_out, int out_size, void* d_ws, size_t ws_size,
                              hipStream_t stream) {
    const float* status   = (const float*)d_in[0];
    const float* rois     = (const float*)d_in[1];
    const float* bbox     = (const float*)d_in[2];
    const float* w_reduce = (const float*)d_in[3];
    const float* w_q1     = (const float*)d_in[4];
    const float* w_q2     = (const float*)d_in[5];
    const float* w_k1     = (const float*)d_in[6];
    const float* w_v1     = (const float*)d_in[7];
    const float* w_k2     = (const float*)d_in[8];
    const float* w_v2     = (const float*)d_in[9];
    const float* w_c1     = (const float*)d_in[10];
    const float* w_c2     = (const float*)d_in[11];
    const float* gamma    = (const float*)d_in[12];
    const float* beta     = (const float*)d_in[13];
    float* out = (float*)d_out;

    const int W_RED_N = 512 * 2304 * 9;   // 10,616,832
    const int W_KV_N  = 512 * 512 * 9;    //  2,359,296
    const int BBOX_N  = 256 * 2304 * 49;  // 28,901,376

    char* ws = (char*)d_ws;
    size_t off = 0;
    auto take = [&](size_t bytes) -> char* {
        char* p = ws + off;
        off = (off + bytes + 255) & ~(size_t)255;
        return p;
    };

    unsigned short* wbR  = (unsigned short*)take((size_t)W_RED_N * 2);
    unsigned short* wbK1 = (unsigned short*)take((size_t)W_KV_N * 2);
    unsigned short* wbV1 = (unsigned short*)take((size_t)W_KV_N * 2);
    unsigned short* wbK2 = (unsigned short*)take((size_t)W_KV_N * 2);
    unsigned short* wbV2 = (unsigned short*)take((size_t)W_KV_N * 2);
    unsigned short* wbC1 = (unsigned short*)take((size_t)W_KV_N * 2);
    unsigned short* wbC2 = (unsigned short*)take((size_t)W_KV_N * 2);
    unsigned short* bboxb = (unsigned short*)take((size_t)BBOX_N * 2);
    float* featf = (float*)take(FEAT_ELEMS * 4);
    unsigned short* featb = (unsigned short*)take(FEAT_ELEMS * 2);
    float* q1 = (float*)take(FEAT_ELEMS * 4);
    float* q2 = (float*)take(FEAT_ELEMS * 4);
    float* k1 = (float*)take(FEAT_ELEMS * 4);
    float* v1 = (float*)take(FEAT_ELEMS * 4);
    float* k2 = (float*)take(FEAT_ELEMS * 4);
    float* v2 = (float*)take(FEAT_ELEMS * 4);
    float* virt = (float*)take(FEAT_ELEMS * 4);
    unsigned short* hb = (unsigned short*)take(FEAT_ELEMS * 2);

    // weights -> bf16, tap-major
    permute_w_bf16<<<(W_RED_N + 255) / 256, 256, 0, stream>>>(w_reduce, wbR, 2304, W_RED_N);
    permute_w_bf16<<<(W_KV_N + 255) / 256, 256, 0, stream>>>(w_k1, wbK1, 512, W_KV_N);
    permute_w_bf16<<<(W_KV_N + 255) / 256, 256, 0, stream>>>(w_v1, wbV1, 512, W_KV_N);
    permute_w_bf16<<<(W_KV_N + 255) / 256, 256, 0, stream>>>(w_k2, wbK2, 512, W_KV_N);
    permute_w_bf16<<<(W_KV_N + 255) / 256, 256, 0, stream>>>(w_v2, wbV2, 512, W_KV_N);
    permute_w_bf16<<<(W_KV_N + 255) / 256, 256, 0, stream>>>(w_c1, wbC1, 512, W_KV_N);
    permute_w_bf16<<<(W_KV_N + 255) / 256, 256, 0, stream>>>(w_c2, wbC2, 512, W_KV_N);
    // bbox_feat -> bf16 channel-last
    permute_cl_bf16<<<(BBOX_N + 255) / 256, 256, 0, stream>>>(bbox, bboxb, 2304, BBOX_N);

    const dim3 convGrid(M_TOT / 128, HIDC / 64);  // (98, 8)

    // feat = conv(bbox, w_reduce): f32 NCHW + bf16 channel-last copies
    conv3x3_wmma<2304><<<convGrid, 256, 0, stream>>>(bboxb, wbR, nullptr, featf, featb);

    status_conv<<<(NB * HIDC * HW + 255) / 256, 256, 0, stream>>>(status, 0, rois, w_q1, q1);
    status_conv<<<(NB * HIDC * HW + 255) / 256, 256, 0, stream>>>(status, 1, rois, w_q2, q2);

    conv3x3_wmma<512><<<convGrid, 256, 0, stream>>>(featb, wbK1, nullptr, k1, nullptr);
    conv3x3_wmma<512><<<convGrid, 256, 0, stream>>>(featb, wbV1, nullptr, v1, nullptr);
    conv3x3_wmma<512><<<convGrid, 256, 0, stream>>>(featb, wbK2, nullptr, k2, nullptr);
    conv3x3_wmma<512><<<convGrid, 256, 0, stream>>>(featb, wbV2, nullptr, v2, nullptr);

    // ---- branch 1
    attention_kernel<<<dim3(HW, NGRP), 256, 0, stream>>>(q1, k1, v1, virt);
    groupnorm_relu<<<NB, 256, 0, stream>>>(virt, gamma, beta, hb);
    conv3x3_wmma<512><<<convGrid, 256, 0, stream>>>(hb, wbC1, featf, out, nullptr);

    // ---- branch 2
    attention_kernel<<<dim3(HW, NGRP), 256, 0, stream>>>(q2, k2, v2, virt);
    groupnorm_relu<<<NB, 256, 0, stream>>>(virt, gamma, beta, hb);
    conv3x3_wmma<512><<<convGrid, 256, 0, stream>>>(hb, wbC2, featf, out + FEAT_ELEMS, nullptr);
}
